// GCN_2473901162738
// MI455X (gfx1250) — matrix operand order repaired and checked by reading, not compiled, not run on hardware
//
#include <hip/hip_runtime.h>

// ---------------------------------------------------------------------------
// GCN on MI455X (gfx1250, wave32, WMMA bf16):
//   layer: H = (X .* norm) @ W      (v_wmma_f32_16x16x32_bf16, f32 accumulate)
//          X' = relu( segsum_edges(g * H[col]) .* norm + b )
// All fp32->bf16 conversion hoisted to pre-passes; GEMM inner loop is pure
// global_load_b128 + wmma (A pre-scaled row-major, B pre-transposed).
// ---------------------------------------------------------------------------

typedef __attribute__((ext_vector_type(16))) __bf16 bfvec16;
typedef __attribute__((ext_vector_type(8)))  float  fvec8;

__device__ __forceinline__ unsigned short f2bf(float f) {
  // round-to-nearest-even f32 -> bf16 (bit pattern)
  unsigned int u = __float_as_uint(f);
  unsigned int r = u + 0x7FFFu + ((u >> 16) & 1u);
  return (unsigned short)(r >> 16);
}

// --------------------------- pre-pass: A = bf16(X .* norm) ------------------
// one float4 -> one uint2 (4 packed bf16) per thread
__global__ void __launch_bounds__(256)
cvt_x(const float4* __restrict__ X4, const float* __restrict__ norm,
      uint2* __restrict__ out, int total4, int kdiv4) {
  const int i = blockIdx.x * blockDim.x + threadIdx.x;
  if (i >= total4) return;
  const float s = norm[i / kdiv4];
  const float4 v = X4[i];
  uint2 o;
  o.x = (unsigned)f2bf(v.x * s) | ((unsigned)f2bf(v.y * s) << 16);
  o.y = (unsigned)f2bf(v.z * s) | ((unsigned)f2bf(v.w * s) << 16);
  out[i] = o;
}

// --------------------------- pre-pass: Wt = bf16(W^T) -----------------------
template <int K, int NC>
__global__ void __launch_bounds__(256)
cvt_w(const float* __restrict__ W, unsigned short* __restrict__ Wt) {
  const int i = blockIdx.x * blockDim.x + threadIdx.x;  // over K*NC
  if (i >= K * NC) return;
  const int n = i / K, k = i % K;
  Wt[i] = f2bf(W[(size_t)k * NC + n]);   // Wt[n][k] = W[k][n]
}

// --------------------------- register-resident WMMA GEMM --------------------
// H[m0:m0+16, :NC] = Abf[m0:m0+16, :K] @ Wt^T ; wave = one 16x16 tile.
// A frag: row m, K contiguous (K 0-15 / 16-31 per lane half).
// B frag: col n, K contiguous (K 0-15 / 16-31 per lane half).
template <int K, int NC>
__global__ void __launch_bounds__(32 * (NC / 16))
gcn_gemm(const unsigned short* __restrict__ Abf,  // [rows][K] bf16, pre-scaled
         const unsigned short* __restrict__ Wt,   // [NC][K]   bf16
         float* __restrict__ H) {
  const int tid  = threadIdx.x;
  const int wave = tid >> 5;
  const int lane = tid & 31;
  const int l15  = lane & 15;
  const int half = lane >> 4;
  const int m0   = blockIdx.x * 16;
  const int n0   = wave * 16;

  const unsigned short* arow = Abf + (size_t)(m0 + l15) * K + half * 16;
  const unsigned short* bcol = Wt  + (size_t)(n0 + l15) * K + half * 16;

  fvec8 acc = {};
#pragma unroll
  for (int kk = 0; kk < K; kk += 32) {
    union { bfvec16 v; uint4 q[2]; } a, b;
    a.q[0] = *(const uint4*)(arow + kk);        // K = kk + 16*half .. +7
    a.q[1] = *(const uint4*)(arow + kk + 8);    // K = kk + 16*half + 8 .. +15
    b.q[0] = *(const uint4*)(bcol + kk);        // K = kk + 16*half .. +7
    b.q[1] = *(const uint4*)(bcol + kk + 8);    // K = kk + 16*half + 8 .. +15
    acc = __builtin_amdgcn_wmma_f32_16x16x32_bf16(
        /*neg_a=*/false, a.v, /*neg_b=*/false, b.v,
        /*c_mod=*/(short)0, acc, /*reuse_a=*/false, /*reuse_b=*/false);
  }
  // D layout: reg r -> row m0 + 8*half + r, col n0 + l15
#pragma unroll
  for (int r = 0; r < 8; ++r)
    H[(size_t)(m0 + half * 8 + r) * NC + n0 + l15] = acc[r];
}

// --------------------------- SpMM layer 1 (fused epilogue) ------------------
// one wave per node; lane owns 4 of 128 channels (float4 gathers).
// writes bf16( relu(agg*norm + b1) * norm )  == layer-2 A, pre-scaled.
__global__ void __launch_bounds__(256)
gcn_spmm1(const float* __restrict__ H, const float* __restrict__ g,
          const int* __restrict__ idx, const float* __restrict__ norm,
          const float* __restrict__ b, uint2* __restrict__ outbf, int n_nodes) {
  const int wave = threadIdx.x >> 5;
  const int lane = threadIdx.x & 31;
  const long r   = (long)blockIdx.x * 8 + wave;
  if (r >= n_nodes) return;
  const long ebase = r * 16;
  const int   my_i = idx[ebase + (lane & 15)];
  const float my_g = g[ebase + (lane & 15)];
  const int c4 = lane * 4;

  float4 acc = {0.f, 0.f, 0.f, 0.f};
#pragma unroll
  for (int e = 0; e < 16; ++e) {
    const int   col = __shfl(my_i, e, 32);
    const float gv  = __shfl(my_g, e, 32);
    const float4 hv = *(const float4*)(H + (size_t)col * 128 + c4);
    acc.x += gv * hv.x; acc.y += gv * hv.y;
    acc.z += gv * hv.z; acc.w += gv * hv.w;
  }
  const float  nr = norm[r];
  const float4 bv = *(const float4*)(b + c4);
  const float o0 = fmaxf(acc.x * nr + bv.x, 0.f) * nr;
  const float o1 = fmaxf(acc.y * nr + bv.y, 0.f) * nr;
  const float o2 = fmaxf(acc.z * nr + bv.z, 0.f) * nr;
  const float o3 = fmaxf(acc.w * nr + bv.w, 0.f) * nr;
  uint2 p;
  p.x = (unsigned)f2bf(o0) | ((unsigned)f2bf(o1) << 16);
  p.y = (unsigned)f2bf(o2) | ((unsigned)f2bf(o3) << 16);
  outbf[(size_t)r * 32 + lane] = p;   // 128 bf16 per row == 32 uint2
}

// --------------------------- SpMM layer 2 (final) ---------------------------
// one wave per node; lane == channel (CLS == 32 == wave32).
__global__ void __launch_bounds__(256)
gcn_spmm2(const float* __restrict__ H, const float* __restrict__ g,
          const int* __restrict__ idx, const float* __restrict__ norm,
          const float* __restrict__ b, float* __restrict__ out, int n_nodes) {
  const int wave = threadIdx.x >> 5;
  const int lane = threadIdx.x & 31;
  const long r   = (long)blockIdx.x * 8 + wave;
  if (r >= n_nodes) return;
  const long ebase = r * 16;
  const int   my_i = idx[ebase + (lane & 15)];
  const float my_g = g[ebase + (lane & 15)];

  float acc = 0.f;
#pragma unroll
  for (int e = 0; e < 16; ++e) {
    const int   col = __shfl(my_i, e, 32);
    const float gv  = __shfl(my_g, e, 32);
    acc += gv * H[(size_t)col * 32 + lane];
  }
  out[(size_t)r * 32 + lane] = acc * norm[r] + b[lane];
}

// ---------------------------------------------------------------------------
extern "C" void kernel_launch(void* const* d_in, const int* in_sizes, int n_in,
                              void* d_out, int out_size, void* d_ws, size_t ws_size,
                              hipStream_t stream) {
  constexpr int N = 100000, F_IN = 256, HID = 128, CLS = 32;

  const float* features = (const float*)d_in[0];
  const float* g_data   = (const float*)d_in[1];
  const float* norm     = (const float*)d_in[2];
  const float* W1       = (const float*)d_in[3];
  const float* b1       = (const float*)d_in[4];
  const float* W2       = (const float*)d_in[5];
  const float* b2       = (const float*)d_in[6];
  const int*   indices  = (const int*)d_in[7];
  // d_in[8] (row_ids) unused: fixed out-degree of 16 by construction.

  // workspace layout (regions reused once their producer/consumer retire):
  //   [0,   51.2MB): Xbf [N,256] bf16   -> reused by X2bf [N,128] bf16
  //   [51.2,102.4MB): H1 [N,128] f32    -> reused by H2 [N,32] f32
  //   [102.4MB, ..): Wt1 (64KB), Wt2 (8KB)
  char* ws = (char*)d_ws;
  unsigned short* Xbf  = (unsigned short*)ws;                       // & X2bf
  float*          H1   = (float*)(ws + (size_t)N * F_IN * 2);      // & H2
  unsigned short* Wt1  = (unsigned short*)(ws + (size_t)N * F_IN * 2 +
                                                (size_t)N * HID * 4);
  unsigned short* Wt2  = Wt1 + (size_t)F_IN * HID;
  unsigned short* X2bf = Xbf;
  float*          H2   = H1;

  // pre-passes: bf16 weights (transposed) + bf16 pre-scaled features
  cvt_w<F_IN, HID><<<(F_IN * HID + 255) / 256, 256, 0, stream>>>(W1, Wt1);
  cvt_w<HID, CLS><<<(HID * CLS + 255) / 256, 256, 0, stream>>>(W2, Wt2);
  cvt_x<<<(N * F_IN / 4 + 255) / 256, 256, 0, stream>>>(
      (const float4*)features, norm, (uint2*)Xbf, N * F_IN / 4, F_IN / 4);

  // Layer 1
  gcn_gemm<F_IN, HID><<<N / 16, 256, 0, stream>>>(Xbf, Wt1, H1);
  gcn_spmm1<<<N / 8, 256, 0, stream>>>(H1, g_data, indices, norm, b1,
                                       (uint2*)X2bf, N);
  // Layer 2
  gcn_gemm<HID, CLS><<<N / 16, 64, 0, stream>>>(X2bf, Wt2, H2);
  gcn_spmm2<<<N / 8, 256, 0, stream>>>(H2, g_data, indices, norm, b2,
                                       (float*)d_out, N);
}
